// DSSL_10376640987772
// MI455X (gfx1250) — compile-verified, matrix operand order checked by reading, not run on hardware
//
#include <hip/hip_runtime.h>
#include <math.h>

// ---------------------------------------------------------------------------
// MI455X (gfx1250) implementation: bf16 WMMA everywhere, fp32 accumulate.
// ---------------------------------------------------------------------------

typedef __attribute__((ext_vector_type(16))) __bf16 v16bf;
typedef __attribute__((ext_vector_type(8)))  float  v8f;

#define BM 128
#define BN 128
#define BK 32

union FragU { uint4 q[2]; v16bf v; };

// Build a 16-half WMMA fragment from two 16-byte chunks.
// Per CDNA5 ISA 16-bit A/B layout: lane holds K = {kb..kb+7} U {kb+16..kb+23},
// kb = 8*(lane>=16); pairs are contiguous in K so each chunk is one uint4.
__device__ inline v16bf frag_from(const __bf16* p0, const __bf16* p1) {
  FragU u;
  u.q[0] = *(const uint4*)p0;
  u.q[1] = *(const uint4*)p1;
  return u.v;
}

// ---------------------------------------------------------------------------
// Generic tiled WMMA GEMM: D = act(A @ B^T(n x k) + bias [+ extras])
// ACT: 0 none, 1 relu, 2 tanh.  OBS_A: A is fp32 obs scaled by invnorm.
// FV: add Wn1[100+feature[m/64]][n].  RESID: add resid[m*N+n].
// SPLITK: atomically accumulate raw fp32 into Df32 (no bias/act).
// ---------------------------------------------------------------------------
template<int ACT, bool OBS_A, bool FV, bool RESID, bool SPLITK>
__global__ void __launch_bounds__(256) gemm_wmma(
    const void* __restrict__ Aptr, const __bf16* __restrict__ Bt,
    const float* __restrict__ bias,
    __bf16* __restrict__ Dbf, float* __restrict__ Df32,
    const float* __restrict__ scaleA,
    const float* __restrict__ resid,
    const float* __restrict__ fvW, const int* __restrict__ feat,
    int M, int N, int K, int lda, int ldbT, int lddbf, int lddf,
    int kChunks, int Vdim)
{
  __shared__ __bf16 As[BM * BK];
  __shared__ __bf16 Bs[BN * BK];

  const int tid  = threadIdx.x;
  const int lane = tid & 31, wave = tid >> 5;
  const int wm = wave >> 2, wn = wave & 3;        // 2 x 4 wave grid
  const int l16 = lane & 15, half = lane >> 4;
  const int kb = half * 8;
  const int mb0 = blockIdx.y * BM, nb0 = blockIdx.x * BN;
  const int ksteps = (K + BK - 1) / BK;
  const int kbeg = blockIdx.z * kChunks;
  const int kend = min(kbeg + kChunks, ksteps);

  v8f acc[4][2];
  #pragma unroll
  for (int i = 0; i < 4; ++i)
    #pragma unroll
    for (int j = 0; j < 2; ++j)
      #pragma unroll
      for (int e = 0; e < 8; ++e) acc[i][j][e] = 0.f;

  const int arow = tid >> 3;          // 0..31
  const int akk  = (tid & 7) * 4;     // 0,4,..,28

  for (int ks = kbeg; ks < kend; ++ks) {
    const int k0 = ks * BK;
    // ---- fill A tile (BM x BK) ----
    #pragma unroll
    for (int r = 0; r < 4; ++r) {
      const int row = r * 32 + arow;
      const int m = mb0 + row;
      #pragma unroll
      for (int e = 0; e < 4; ++e) {
        const int kg = k0 + akk + e;
        float x = 0.f;
        if (m < M && kg < K) {
          if constexpr (OBS_A) {
            x = ((const float*)Aptr)[(size_t)m * lda + kg] *
                scaleA[(size_t)(m >> 6) * Vdim + kg];
          } else {
            x = (float)((const __bf16*)Aptr)[(size_t)m * lda + kg];
          }
        }
        As[row * BK + akk + e] = (__bf16)x;
      }
      if (r == 0 && ks + 1 < kend) {   // global_prefetch_b8 of next k-tile
        const int kg2 = k0 + BK + akk;
        if (m < M && kg2 < K) {
          if constexpr (OBS_A)
            __builtin_prefetch(&((const float*)Aptr)[(size_t)m * lda + kg2], 0, 0);
          else
            __builtin_prefetch(&((const __bf16*)Aptr)[(size_t)m * lda + kg2], 0, 0);
        }
      }
    }
    // ---- fill B tile (BN x BK), Bt is N x K row-major ----
    #pragma unroll
    for (int r = 0; r < 4; ++r) {
      const int nrow = r * 32 + arow;
      const int n = nb0 + nrow;
      #pragma unroll
      for (int e = 0; e < 4; ++e) {
        const int kg = k0 + akk + e;
        float x = 0.f;
        if (n < N && kg < K) x = (float)Bt[(size_t)n * ldbT + kg];
        Bs[nrow * BK + akk + e] = (__bf16)x;
      }
      if (r == 0 && ks + 1 < kend) {
        const int kg2 = k0 + BK + akk;
        if (n < N && kg2 < K) __builtin_prefetch(&Bt[(size_t)n * ldbT + kg2], 0, 0);
      }
    }
    __syncthreads();

    v16bf afr[4], bfr[2];
    #pragma unroll
    for (int mt = 0; mt < 4; ++mt) {
      const int row = wm * 64 + mt * 16 + l16;
      afr[mt] = frag_from(&As[row * BK + kb], &As[row * BK + kb + 16]);
    }
    #pragma unroll
    for (int nt = 0; nt < 2; ++nt) {
      const int nrow = wn * 32 + nt * 16 + l16;
      bfr[nt] = frag_from(&Bs[nrow * BK + kb], &Bs[nrow * BK + kb + 16]);
    }
    #pragma unroll
    for (int mt = 0; mt < 4; ++mt)
      #pragma unroll
      for (int nt = 0; nt < 2; ++nt)
        acc[mt][nt] = __builtin_amdgcn_wmma_f32_16x16x32_bf16(
            false, afr[mt], false, bfr[nt], (short)0, acc[mt][nt], false, false);
    __syncthreads();
  }

  // ---- epilogue ----
  #pragma unroll
  for (int mt = 0; mt < 4; ++mt) {
    #pragma unroll
    for (int nt = 0; nt < 2; ++nt) {
      const int n = nb0 + wn * 32 + nt * 16 + l16;
      #pragma unroll
      for (int r = 0; r < 8; ++r) {
        const int m = mb0 + wm * 64 + mt * 16 + half * 8 + r;
        if (m < M && n < N) {
          float v = acc[mt][nt][r];
          if constexpr (SPLITK) {
            atomicAdd(&Df32[(size_t)m * lddf + n], v);
          } else {
            if (bias) v += bias[n];
            if constexpr (FV)
              v += fvW[(size_t)(100 + feat[m >> 6]) * 100 + n];
            if constexpr (RESID)
              v += resid[(size_t)m * N + n];
            if constexpr (ACT == 1) v = fmaxf(v, 0.f);
            if constexpr (ACT == 2) v = tanhf(v);
            if (Dbf)  Dbf[(size_t)m * lddbf + n] = (__bf16)v;
            if (Df32) Df32[(size_t)m * lddf + n] = v;
          }
        }
      }
    }
  }
}

// ---------------------------------------------------------------------------
// Edge kernel: one block per (batch b, source node i).
// e1[j] = relu(P[b,i] + Q[b,j]); e2 = relu(LN(e1 @ We2 + be2));
// agg[b,i] = (sum_{j!=i} e2[j]) @ We3 + 63*be3.
// ---------------------------------------------------------------------------
__global__ void __launch_bounds__(128) edge_kernel(
    const float* __restrict__ P, const float* __restrict__ Q,
    const __bf16* __restrict__ We2T, const float* __restrict__ be2,
    const float* __restrict__ lng, const float* __restrict__ lnb,
    const float* __restrict__ We3, const float* __restrict__ be3,
    float* __restrict__ agg)
{
  __shared__ __bf16 e1[64 * 128];
  __shared__ float  stage[64 * 112];
  __shared__ float  s[112];
  __shared__ float  Pi[112];

  const int b = blockIdx.y, i = blockIdx.x;
  const int tid = threadIdx.x;
  const int lane = tid & 31, wave = tid >> 5;
  const int l16 = lane & 15, half = lane >> 4, kb = half * 8;

  if (tid < 112) {
    s[tid]  = 0.f;
    Pi[tid] = (tid < 100) ? P[(size_t)(b * 64 + i) * 100 + tid] : 0.f;
  }
  __syncthreads();

  { // build e1 tile (64 rows j, 128 padded k) in bf16
    const int j = tid >> 1, c0 = (tid & 1) * 64;
    const float* Qj = Q + (size_t)(b * 64 + j) * 100;
    for (int e = 0; e < 64; ++e) {
      const int k = c0 + e;
      float v = 0.f;
      if (k < 100) v = fmaxf(Pi[k] + Qj[k], 0.f);
      e1[j * 128 + k] = (__bf16)v;
    }
  }
  __syncthreads();

  // e2acc = e1(64x128) @ We2T^T : 4 waves x 7 n-tiles, 4 k-steps
  v8f acc[7];
  #pragma unroll
  for (int nt = 0; nt < 7; ++nt)
    #pragma unroll
    for (int e = 0; e < 8; ++e) acc[nt][e] = 0.f;

  #pragma unroll
  for (int ks = 0; ks < 4; ++ks) {
    const int k0 = ks * 32;
    const int row = wave * 16 + l16;
    const v16bf a = frag_from(&e1[row * 128 + k0 + kb],
                              &e1[row * 128 + k0 + kb + 16]);
    #pragma unroll
    for (int nt = 0; nt < 7; ++nt) {
      const int n = nt * 16 + l16;
      const v16bf bf = frag_from(&We2T[(size_t)n * 128 + k0 + kb],
                                 &We2T[(size_t)n * 128 + k0 + kb + 16]);
      acc[nt] = __builtin_amdgcn_wmma_f32_16x16x32_bf16(
          false, a, false, bf, (short)0, acc[nt], false, false);
    }
  }
  #pragma unroll
  for (int nt = 0; nt < 7; ++nt) {
    const int n = nt * 16 + l16;
    #pragma unroll
    for (int r = 0; r < 8; ++r) {
      const int m = wave * 16 + half * 8 + r;
      stage[m * 112 + n] = acc[nt][r];
    }
  }
  __syncthreads();

  // per-row LayerNorm + relu, masked sum over j != i into s[]
  if (tid < 64) {
    const int row = tid;
    float mu = 0.f;
    for (int n = 0; n < 100; ++n) mu += stage[row * 112 + n] + be2[n];
    mu *= 0.01f;
    float var = 0.f;
    for (int n = 0; n < 100; ++n) {
      const float d = stage[row * 112 + n] + be2[n] - mu;
      var += d * d;
    }
    var *= 0.01f;
    const float rinv = rsqrtf(var + 1e-5f);
    if (row != i) {
      for (int n = 0; n < 100; ++n) {
        float v = (stage[row * 112 + n] + be2[n] - mu) * rinv * lng[n] + lnb[n];
        v = fmaxf(v, 0.f);
        atomicAdd(&s[n], v);   // ds_add_f32
      }
    }
  }
  __syncthreads();

  // agg row = s @ We3 + 63*be3
  if (tid < 100) {
    float r = 63.f * be3[tid];
    for (int k = 0; k < 100; ++k) r += s[k] * We3[(size_t)k * 100 + tid];
    agg[(size_t)(b * 64 + i) * 100 + tid] = r;
  }
}

// ---------------------------------------------------------------------------
// Helper kernels
// ---------------------------------------------------------------------------
__global__ void zero_kernel(uint4* p, size_t n) {
  size_t i = (size_t)blockIdx.x * 256 + threadIdx.x;
  const size_t stride = (size_t)gridDim.x * 256;
  uint4 z; z.x = z.y = z.z = z.w = 0u;
  for (; i < n; i += stride) p[i] = z;
}

__global__ void colnorm_kernel(const float* __restrict__ obs,
                               float* __restrict__ invnorm, int V) {
  const int v = blockIdx.x * 128 + threadIdx.x;
  const int b = blockIdx.y;
  if (v >= V) return;
  float sum = 0.f;
  for (int n = 0; n < 64; ++n) {
    const float x = obs[((size_t)b * 64 + n) * V + v];
    sum += x * x;
  }
  invnorm[(size_t)b * V + v] = 1.f / fmaxf(sqrtf(sum), 1e-6f);
}

// Tiled transpose + fp32 -> bf16: out[n][kout+k] = W[rowoff+k][n]
__global__ void cvtT_kernel(const float* __restrict__ W, int ldw, int rowoff,
                            __bf16* __restrict__ out, int ldout, int kout,
                            int Nv, int Kv) {
  __shared__ float t[32][33];
  const int k0 = blockIdx.x * 32, n0 = blockIdx.y * 32;
  #pragma unroll
  for (int r = 0; r < 4; ++r) {
    const int k = k0 + threadIdx.y + r * 8;
    const int n = n0 + threadIdx.x;
    if (k < Kv && n < Nv)
      t[threadIdx.y + r * 8][threadIdx.x] = W[(size_t)(rowoff + k) * ldw + n];
  }
  __syncthreads();
  #pragma unroll
  for (int r = 0; r < 4; ++r) {
    const int n = n0 + threadIdx.y + r * 8;
    const int k = k0 + threadIdx.x;
    if (k < Kv && n < Nv)
      out[(size_t)n * ldout + kout + k] = (__bf16)t[threadIdx.x][threadIdx.y + r * 8];
  }
}

__global__ void bias_tanh_kernel(const float* __restrict__ X,
                                 const float* __restrict__ bias,
                                 __bf16* __restrict__ out,
                                 int M, int N, int ldx, int ldo) {
  const size_t idx = (size_t)blockIdx.x * 256 + threadIdx.x;
  if (idx >= (size_t)M * N) return;
  const int m = (int)(idx / N), n = (int)(idx % N);
  out[(size_t)m * ldo + n] = (__bf16)tanhf(X[(size_t)m * ldx + n] + bias[n]);
}

__global__ void pack_agg_kernel(const float* __restrict__ agg,
                                __bf16* __restrict__ nodeagg) {
  const int idx = blockIdx.x * 256 + threadIdx.x;
  if (idx >= 4096 * 100) return;
  const int m = idx / 100, n = idx % 100;
  nodeagg[(size_t)m * 200 + 100 + n] = (__bf16)agg[idx];
}

// wave-per-row LayerNorm (+bias folded in) + relu -> bf16
__global__ void __launch_bounds__(256) ln_act_kernel(
    const float* __restrict__ X, const float* __restrict__ bias,
    const float* __restrict__ g, const float* __restrict__ bb,
    __bf16* __restrict__ out, int M, int N, int ldx, int ldo) {
  const int wave = threadIdx.x >> 5, lane = threadIdx.x & 31;
  const int row = blockIdx.x * 8 + wave;
  if (row >= M) return;
  float x[4]; float sum = 0.f;
  #pragma unroll
  for (int i = 0; i < 4; ++i) {
    const int c = lane + i * 32;
    float v = 0.f;
    if (c < N) v = X[(size_t)row * ldx + c] + (bias ? bias[c] : 0.f);
    x[i] = v; sum += v;
  }
  #pragma unroll
  for (int m = 16; m >= 1; m >>= 1) sum += __shfl_xor(sum, m, 32);
  const float mu = sum / N;
  float vs = 0.f;
  #pragma unroll
  for (int i = 0; i < 4; ++i) {
    const int c = lane + i * 32;
    if (c < N) { const float d = x[i] - mu; vs += d * d; }
  }
  #pragma unroll
  for (int m = 16; m >= 1; m >>= 1) vs += __shfl_xor(vs, m, 32);
  const float rinv = rsqrtf(vs / N + 1e-5f);
  #pragma unroll
  for (int i = 0; i < 4; ++i) {
    const int c = lane + i * 32;
    if (c < N) {
      const float v = (x[i] - mu) * rinv * g[c] + bb[c];
      out[(size_t)row * ldo + c] = (__bf16)fmaxf(v, 0.f);
    }
  }
}

// ---------------------------------------------------------------------------
// Host orchestration
// ---------------------------------------------------------------------------
extern "C" void kernel_launch(void* const* d_in, const int* in_sizes, int n_in,
                              void* d_out, int out_size, void* d_ws, size_t ws_size,
                              hipStream_t stream) {
  (void)in_sizes; (void)n_in; (void)out_size; (void)ws_size;
  const float* obs  = (const float*)d_in[0];
  const int*   feat = (const int*)  d_in[1];
  const float* Wq0  = (const float*)d_in[2];
  const float* bq0  = (const float*)d_in[3];
  const float* Wq1  = (const float*)d_in[4];
  const float* bq1  = (const float*)d_in[5];
  const float* We1  = (const float*)d_in[6];
  const float* be1  = (const float*)d_in[7];
  const float* We2  = (const float*)d_in[8];
  const float* be2  = (const float*)d_in[9];
  const float* lneg = (const float*)d_in[10];
  const float* lneb = (const float*)d_in[11];
  const float* We3  = (const float*)d_in[12];
  const float* be3  = (const float*)d_in[13];
  const float* Wn1  = (const float*)d_in[14];
  const float* bn1  = (const float*)d_in[15];
  const float* Wn2  = (const float*)d_in[16];
  const float* bn2  = (const float*)d_in[17];
  const float* lnng = (const float*)d_in[18];
  const float* lnnb = (const float*)d_in[19];
  const float* Wn3  = (const float*)d_in[20];
  const float* bn3  = (const float*)d_in[21];
  const float* Wp0  = (const float*)d_in[22];
  const float* bp0  = (const float*)d_in[23];
  const float* Wp1  = (const float*)d_in[24];
  const float* bp1  = (const float*)d_in[25];
  float* out = (float*)d_out;

  char* base = (char*)d_ws; size_t off = 0;
  auto alloc = [&](size_t bytes) -> void* {
    void* p = base + off; off = (off + bytes + 255) & ~(size_t)255; return p;
  };
  float*  invn   = (float*) alloc((size_t)64 * 32000 * 4);
  __bf16* Wq0T   = (__bf16*)alloc((size_t)600 * 32000 * 2);
  __bf16* Wp1T   = (__bf16*)alloc((size_t)32000 * 600 * 2);
  __bf16* Wq1T   = (__bf16*)alloc((size_t)100 * 600 * 2);
  __bf16* We1Pt  = (__bf16*)alloc((size_t)100 * 100 * 2);
  __bf16* We1Qt  = (__bf16*)alloc((size_t)100 * 100 * 2);
  __bf16* We2T   = (__bf16*)alloc((size_t)112 * 128 * 2);   // zero-padded
  __bf16* Wn1T   = (__bf16*)alloc((size_t)100 * 200 * 2);
  __bf16* Wn2T   = (__bf16*)alloc((size_t)100 * 100 * 2);
  __bf16* Wn3T   = (__bf16*)alloc((size_t)100 * 100 * 2);
  __bf16* Wp0T   = (__bf16*)alloc((size_t)600 * 100 * 2);
  float*  h1f    = (float*) alloc((size_t)4096 * 600 * 4);  // split-K target
  __bf16* h1     = (__bf16*)alloc((size_t)4096 * 600 * 2);
  float*  z      = (float*) alloc((size_t)4096 * 100 * 4);
  __bf16* nodeag = (__bf16*)alloc((size_t)4096 * 200 * 2);  // [node | agg]
  float*  Pbuf   = (float*) alloc((size_t)4096 * 100 * 4);
  float*  Qbuf   = (float*) alloc((size_t)4096 * 100 * 4);
  float*  aggb   = (float*) alloc((size_t)4096 * 100 * 4);
  __bf16* nh1    = (__bf16*)alloc((size_t)4096 * 100 * 2);
  float*  nh2f   = (float*) alloc((size_t)4096 * 100 * 4);
  __bf16* nh2    = (__bf16*)alloc((size_t)4096 * 100 * 2);
  __bf16* nexts  = (__bf16*)alloc((size_t)4096 * 100 * 2);
  __bf16* tbuf   = (__bf16*)alloc((size_t)4096 * 600 * 2);
  const size_t total = off;

  zero_kernel<<<2048, 256, 0, stream>>>((uint4*)d_ws, total / 16);
  colnorm_kernel<<<dim3(250, 64), 128, 0, stream>>>(obs, invn, 32000);

  const dim3 tb32(32, 8);
  cvtT_kernel<<<dim3(1000, 19), tb32, 0, stream>>>(Wq0, 600,   0,    Wq0T, 32000, 0,   600,   32000);
  cvtT_kernel<<<dim3(19, 1000), tb32, 0, stream>>>(Wp1, 32000, 0,    Wp1T, 600,   0,   32000, 600);
  cvtT_kernel<<<dim3(19, 4),    tb32, 0, stream>>>(Wq1, 200,   0,    Wq1T, 600,   0,   100,   600);
  cvtT_kernel<<<dim3(4, 4),     tb32, 0, stream>>>(We1, 100,   0,    We1Pt, 100,  0,   100,   100);
  cvtT_kernel<<<dim3(4, 4),     tb32, 0, stream>>>(We1, 100,   100,  We1Qt, 100,  0,   100,   100);
  cvtT_kernel<<<dim3(4, 4),     tb32, 0, stream>>>(We2, 100,   0,    We2T, 128,   0,   100,   100);
  cvtT_kernel<<<dim3(4, 4),     tb32, 0, stream>>>(Wn1, 100,   0,    Wn1T, 200,   0,   100,   100);
  cvtT_kernel<<<dim3(4, 4),     tb32, 0, stream>>>(Wn1, 100,   1100, Wn1T, 200,   100, 100,   100);
  cvtT_kernel<<<dim3(4, 4),     tb32, 0, stream>>>(Wn2, 100,   0,    Wn2T, 100,   0,   100,   100);
  cvtT_kernel<<<dim3(4, 4),     tb32, 0, stream>>>(Wn3, 100,   0,    Wn3T, 100,   0,   100,   100);
  cvtT_kernel<<<dim3(4, 19),    tb32, 0, stream>>>(Wp0, 600,   0,    Wp0T, 100,   0,   600,   100);

  // G1: h1f += norm(obs) @ Wq0   (split-K x4, atomic fp32)
  gemm_wmma<0, true, false, false, true><<<dim3(5, 32, 4), 256, 0, stream>>>(
      obs, Wq0T, nullptr, nullptr, h1f, invn, nullptr, nullptr, nullptr,
      4096, 600, 32000, 32000, 32000, 0, 600, 250, 32000);
  bias_tanh_kernel<<<(4096 * 600 + 255) / 256, 256, 0, stream>>>(
      h1f, bq0, h1, 4096, 600, 600, 600);

  // G2: node/z = h1 @ Wq1[:, :100] + bq1
  gemm_wmma<0, false, false, false, false><<<dim3(1, 32, 1), 256, 0, stream>>>(
      h1, Wq1T, bq1, nodeag, z, nullptr, nullptr, nullptr, nullptr,
      4096, 100, 600, 600, 600, 200, 100, 19, 0);

  // G3/G4: P = node @ We1_top + be1 ; Q = node @ We1_bot
  gemm_wmma<0, false, false, false, false><<<dim3(1, 32, 1), 256, 0, stream>>>(
      nodeag, We1Pt, be1, nullptr, Pbuf, nullptr, nullptr, nullptr, nullptr,
      4096, 100, 100, 200, 100, 0, 100, 4, 0);
  gemm_wmma<0, false, false, false, false><<<dim3(1, 32, 1), 256, 0, stream>>>(
      nodeag, We1Qt, nullptr, nullptr, Qbuf, nullptr, nullptr, nullptr, nullptr,
      4096, 100, 100, 200, 100, 0, 100, 4, 0);

  // Edge MLP + aggregation
  edge_kernel<<<dim3(64, 64), 128, 0, stream>>>(
      Pbuf, Qbuf, We2T, be2, lneg, lneb, We3, be3, aggb);
  pack_agg_kernel<<<1600, 256, 0, stream>>>(aggb, nodeag);

  // G7: nh1 = relu([node|agg] @ Wn1_sel + bn1 + Wn1[100+feat])
  gemm_wmma<1, false, true, false, false><<<dim3(1, 32, 1), 256, 0, stream>>>(
      nodeag, Wn1T, bn1, nh1, nullptr, nullptr, nullptr, Wn1, feat,
      4096, 100, 200, 200, 200, 100, 0, 7, 0);

  // G8: nh2 = relu(LN(nh1 @ Wn2 + bn2))
  gemm_wmma<0, false, false, false, false><<<dim3(1, 32, 1), 256, 0, stream>>>(
      nh1, Wn2T, nullptr, nullptr, nh2f, nullptr, nullptr, nullptr, nullptr,
      4096, 100, 100, 100, 100, 0, 100, 4, 0);
  ln_act_kernel<<<512, 256, 0, stream>>>(nh2f, bn2, lnng, lnnb, nh2, 4096, 100, 100, 100);

  // G9: next = z + nh2 @ Wn3 + bn3
  gemm_wmma<0, false, false, true, false><<<dim3(1, 32, 1), 256, 0, stream>>>(
      nh2, Wn3T, bn3, nexts, nullptr, nullptr, z, nullptr, nullptr,
      4096, 100, 100, 100, 100, 100, 0, 4, 0);

  // G10: t = tanh(next @ Wp0 + bp0)
  gemm_wmma<2, false, false, false, false><<<dim3(5, 32, 1), 256, 0, stream>>>(
      nexts, Wp0T, bp0, tbuf, nullptr, nullptr, nullptr, nullptr, nullptr,
      4096, 600, 100, 100, 100, 600, 0, 4, 0);

  // G11: out = t @ Wp1 + bp1   (fp32 output)
  gemm_wmma<0, false, false, false, false><<<dim3(250, 32, 1), 256, 0, stream>>>(
      tbuf, Wp1T, bp1, nullptr, out, nullptr, nullptr, nullptr, nullptr,
      4096, 32000, 600, 600, 600, 0, 32000, 19, 0);
}
